// DAttentionBaseline_56470230008330
// MI455X (gfx1250) — matrix operand order repaired and checked by reading, not compiled, not run on hardware
//
#include <hip/hip_runtime.h>
#include <hip/hip_bf16.h>

// ---------------------------------------------------------------------------
// CDNA5 (gfx1250) deformable attention forward: bf16 WMMA + TDM pipeline.
// Token-major ([S,256]) layouts so every WMMA fragment K-gather is contiguous.
// ---------------------------------------------------------------------------

typedef __attribute__((ext_vector_type(16))) __bf16          v16bf;
typedef __attribute__((ext_vector_type(8)))  float           v8f;
typedef __attribute__((ext_vector_type(8)))  unsigned short  u16x8;
typedef __attribute__((ext_vector_type(16))) unsigned short  u16x16;
typedef __attribute__((ext_vector_type(4)))  unsigned int    v4u;
typedef __attribute__((ext_vector_type(8)))  int             v8i;
typedef __attribute__((ext_vector_type(4)))  int             v4i;

#define HW   4096
#define CDIM 256
#define NGRP 4
#define NGC  64
#define NHC  32
#define ATTN_SCALE 0.17677669529663687f   // 32^-0.5

#if __has_builtin(__builtin_amdgcn_tensor_load_to_lds) && __has_builtin(__builtin_amdgcn_s_wait_tensorcnt)
#define HAVE_TDM 1
#else
#define HAVE_TDM 0
#endif

union BF16Frag {
    v16bf  v;
    u16x8  h[2];
    u16x16 w;
};

static __device__ __forceinline__ unsigned short f32_to_bf16u(float f) {
    unsigned int u = __builtin_bit_cast(unsigned int, f);
    u += 0x7FFFu + ((u >> 16) & 1u);          // round-to-nearest-even
    return (unsigned short)(u >> 16);
}
static __device__ __forceinline__ float bf16u_to_f32(unsigned short u) {
    unsigned int x = ((unsigned int)u) << 16;
    return __builtin_bit_cast(float, x);
}

// ---------------------------------------------------------------------------
// f32 [256,4096] channel-major -> bf16 [4096,256] token-major
// ---------------------------------------------------------------------------
__global__ void f2bf_tr_kernel(const float* __restrict__ in,
                               unsigned short* __restrict__ outT) {
    const int i = blockIdx.x * blockDim.x + threadIdx.x;
    if (i >= CDIM * HW) return;
    const int c = i >> 12, s = i & 4095;
    outT[(size_t)s * CDIM + c] = f32_to_bf16u(in[i]);
}
// plain f32 -> bf16 (weights)
__global__ void f2bf_kernel(const float* __restrict__ in,
                            unsigned short* __restrict__ out, int n) {
    const int i = blockIdx.x * blockDim.x + threadIdx.x;
    if (i < n) out[i] = f32_to_bf16u(in[i]);
}

// ---------------------------------------------------------------------------
// GEMM: out[o,s] = sum_c W[o,c] * XT[s,c] + bias[o]
// W bf16 [256,256] row-major, XT bf16 [S,256] token-major.
// One wave per 16x16 tile; 8 x v_wmma_f32_16x16x32_bf16.
// Optional outputs: outF f32 [256,S], outT bf16 [S,256], outC bf16 [256,S].
// ---------------------------------------------------------------------------
__launch_bounds__(128)
__global__ void gemm_bias_kernel(const unsigned short* __restrict__ Wb,
                                 const unsigned short* __restrict__ XbT,
                                 const float* __restrict__ bias,
                                 float* __restrict__ outF,
                                 unsigned short* __restrict__ outT,
                                 unsigned short* __restrict__ outC,
                                 int S) {
    const int lane = threadIdx.x & 31;
    const int wave = threadIdx.x >> 5;
    const int tile = blockIdx.x * 4 + wave;
    const int stiles = S >> 4;
    const int o0 = (tile / stiles) << 4;
    const int s0 = (tile % stiles) << 4;
    const int m = lane & 15;
    const int half = lane >> 4;

    v8f acc = {};
    const unsigned short* wr0 = Wb + (o0 + m) * CDIM + half * 16;
    const unsigned short* xr0 = XbT + (size_t)(s0 + m) * CDIM + half * 16;
#pragma unroll
    for (int kk = 0; kk < CDIM; kk += 32) {
        BF16Frag a, b;
        a.w    = *(const u16x16*)(wr0 + kk);       // K {16h..16h+15}, 32B aligned
        b.w    = *(const u16x16*)(xr0 + kk);       // K {16h..16h+15}, 32B aligned
        acc = __builtin_amdgcn_wmma_f32_16x16x32_bf16(
            false, a.v, false, b.v, (short)0, acc, false, false);
    }
    float vals[8];
#pragma unroll
    for (int r = 0; r < 8; ++r) vals[r] = acc[r] + bias[o0 + r + 8 * half];
    if (outF) {
#pragma unroll
        for (int r = 0; r < 8; ++r)
            outF[(size_t)(o0 + r + 8 * half) * S + s0 + m] = vals[r];
    }
    if (outC) {
#pragma unroll
        for (int r = 0; r < 8; ++r)
            outC[(size_t)(o0 + r + 8 * half) * S + s0 + m] = f32_to_bf16u(vals[r]);
    }
    if (outT) {
        u16x8 pk;
#pragma unroll
        for (int r = 0; r < 8; ++r) pk[r] = f32_to_bf16u(vals[r]);
        *(u16x8*)(outT + (size_t)(s0 + m) * CDIM + o0 + 8 * half) = pk;
    }
}

// ---------------------------------------------------------------------------
// 9x9 depthwise conv on q (channel-major f32), groups of 64 channels
// ---------------------------------------------------------------------------
__global__ void dwconv_kernel(const float* __restrict__ qf,
                              const float* __restrict__ wdw,
                              const float* __restrict__ bdw,
                              float* __restrict__ odw) {
    const int idx = blockIdx.x * blockDim.x + threadIdx.x;
    if (idx >= CDIM * HW) return;
    const int hw  = idx & 4095;
    const int gch = idx >> 12;
    const int cg  = gch & 63;
    const int hh  = hw >> 6, ww = hw & 63;
    float acc = bdw[cg];
#pragma unroll
    for (int ky = 0; ky < 9; ++ky) {
        const int iy = hh + ky - 4;
        if (iy < 0 || iy > 63) continue;
#pragma unroll
        for (int kx = 0; kx < 9; ++kx) {
            const int ix = ww + kx - 4;
            if (ix < 0 || ix > 63) continue;
            acc += qf[gch * HW + iy * 64 + ix] * wdw[cg * 81 + ky * 9 + kx];
        }
    }
    odw[idx] = acc;
}

// ---------------------------------------------------------------------------
// Channel LN + exact GELU + 1x1(2,64) + tanh offset + ref grid -> pos (y,x)
// ---------------------------------------------------------------------------
__global__ void lnpw_kernel(const float* __restrict__ odw,
                            const float* __restrict__ ln_g,
                            const float* __restrict__ ln_b,
                            const float* __restrict__ wpw,
                            float* __restrict__ pos) {
    const int idx = blockIdx.x * blockDim.x + threadIdx.x;
    if (idx >= NGRP * HW) return;
    const int g = idx >> 12;
    const int hw = idx & 4095;
    const float* base = odw + (size_t)g * NGC * HW + hw;
    float s = 0.f, s2 = 0.f;
    for (int c = 0; c < NGC; ++c) {
        const float v = base[c * HW];
        s += v; s2 += v * v;
    }
    const float mu = s * (1.0f / NGC);
    const float var = s2 * (1.0f / NGC) - mu * mu;
    const float rstd = rsqrtf(var + 1e-5f);
    float o0 = 0.f, o1 = 0.f;
    for (int c = 0; c < NGC; ++c) {
        const float v = base[c * HW];
        const float t = (v - mu) * rstd * ln_g[c] + ln_b[c];
        const float ge = 0.5f * t * (1.0f + erff(t * 0.70710678118f));
        o0 += wpw[c] * ge;
        o1 += wpw[NGC + c] * ge;
    }
    const int hh = hw >> 6, ww = hw & 63;
    const float ry = ((float)hh + 0.5f) * (2.0f / 64.0f) - 1.0f;
    const float rx = ((float)ww + 0.5f) * (2.0f / 64.0f) - 1.0f;
    pos[idx * 2 + 0] = tanhf(o0) * (2.0f / 64.0f) + ry;   // off_range * 2
    pos[idx * 2 + 1] = tanhf(o1) * (2.0f / 64.0f) + rx;
}

// ---------------------------------------------------------------------------
// Bilinear grid_sample of x (align_corners=True, zeros pad) -> xsT [S,256] bf16
// ---------------------------------------------------------------------------
__global__ void sample_kernel(const float* __restrict__ x,
                              const float* __restrict__ pos,
                              unsigned short* __restrict__ xsT) {
    const int idx = blockIdx.x * blockDim.x + threadIdx.x;
    if (idx >= CDIM * HW) return;
    const int s = idx & 4095;
    const int c = idx >> 12;
    const int g = c >> 6;
    const float py = pos[((size_t)g * HW + s) * 2 + 0];
    const float px = pos[((size_t)g * HW + s) * 2 + 1];
    const float gx = (px + 1.0f) * 0.5f * 63.0f;
    const float gy = (py + 1.0f) * 0.5f * 63.0f;
    const float x0 = floorf(gx), y0 = floorf(gy);
    const float fx = gx - x0, fy = gy - y0;
    const int ix0 = (int)x0, iy0 = (int)y0;
    float out = 0.f;
#pragma unroll
    for (int dy = 0; dy < 2; ++dy) {
        const int iy = iy0 + dy;
        if (iy < 0 || iy > 63) continue;
        const float wy = dy ? fy : (1.0f - fy);
#pragma unroll
        for (int dx = 0; dx < 2; ++dx) {
            const int ix = ix0 + dx;
            if (ix < 0 || ix > 63) continue;
            const float wx = dx ? fx : (1.0f - fx);
            out += wy * wx * x[c * HW + iy * 64 + ix];
        }
    }
    xsT[(size_t)s * CDIM + c] = f32_to_bf16u(out);
}

// ---------------------------------------------------------------------------
// TDM: async 2D bf16 tile (32x32) -> LDS.  D# packed per ISA 8.3/8.4.
// 6-arg toolchain variant: (g0, g1, g2, g3, g4, cpol).
// ---------------------------------------------------------------------------
#if HAVE_TDM
static __device__ __forceinline__ void tdm_tile32x32(unsigned int lds_byte_off,
                                                     const unsigned short* gptr,
                                                     unsigned int stride_elems) {
    const unsigned long long ga = (unsigned long long)(size_t)gptr;
    v4u g0;
    g0[0] = 1u;                                          // count=1, user D#
    g0[1] = lds_byte_off;                                // lds_addr
    g0[2] = (unsigned int)ga;                            // global_addr[31:0]
    g0[3] = (unsigned int)((ga >> 32) & 0x01FFFFFFu) | (2u << 30);  // addr[56:32] | type=2
    v8i g1;
    g1[0] = (int)(1u << 16);                             // data_size = 2 bytes
    g1[1] = (int)((stride_elems & 0xFFFFu) << 16);       // tensor_dim0[15:0]
    g1[2] = (int)((stride_elems >> 16) | (0xFFFFu << 16)); // dim0[31:16] | dim1[15:0]
    g1[3] = (int)(32u << 16);                            // dim1[31:16]=0 | tile_dim0=32
    g1[4] = 32;                                          // tile_dim1=32, tile_dim2=0
    g1[5] = (int)stride_elems;                           // tensor_dim0_stride[31:0]
    g1[6] = 0;
    g1[7] = 0;
    const v4i z4 = {0, 0, 0, 0};
    const v8i z8 = {0, 0, 0, 0, 0, 0, 0, 0};
    __builtin_amdgcn_tensor_load_to_lds(g0, g1, z4, z4, z8, 0);
}
#endif

// ---------------------------------------------------------------------------
// Fused flash attention per head.  grid=(64,8), block=128 (4 waves).
// S = Qt*K via WMMA + LDS RPE bilinear bias, online softmax, O += P*Vt.
// K/V 32x32 tiles staged in LDS (TDM double-buffered when available).
// LDS map (u16 units): [0,16256) rpe bf16 (stride 128) | [16256,+2048) P
//                      | [18304,+2048) K bufs | [20352,+2048) V bufs
// ---------------------------------------------------------------------------
#define RPE_OFF 0
#define P_OFF   16256
#define K_OFF   18304
#define V_OFF   20352
#define SMEM_U16 22400

__launch_bounds__(128)
__global__ void attn_kernel(const unsigned short* __restrict__ qT,   // [HW,256]
                            const unsigned short* __restrict__ kT,   // [HW,256]
                            const unsigned short* __restrict__ vC,   // [256,HW]
                            const float* __restrict__ pos,           // [4][HW][2]
                            const float* __restrict__ rpe,           // [8][127][127]
                            unsigned short* __restrict__ attnT) {    // [HW,256]
    __shared__ unsigned short smem[SMEM_U16];

    const int h = blockIdx.y;
    const int g = h >> 1;
    const int wave = threadIdx.x >> 5;
    const int lane = threadIdx.x & 31;
    const int m = lane & 15;
    const int half = lane >> 4;
    const int mbase = blockIdx.x * 64 + wave * 16;

    for (int idx = threadIdx.x; idx < 127 * 127; idx += 128) {
        const int r = idx / 127, c = idx % 127;
        smem[RPE_OFF + r * 128 + c] = f32_to_bf16u(rpe[(h * 127 + r) * 127 + c]);
    }
    __syncthreads();

    // Q^T fragment (A layout): contiguous K gather from token-major qT
    BF16Frag aq;
    {
        const unsigned short* qr = qT + (size_t)(mbase + m) * CDIM + h * NHC + half * 16;
        aq.w = *(const u16x16*)(qr);
    }

    float qy[8], qx[8], mmax[8], lsum[8];
#pragma unroll
    for (int r = 0; r < 8; ++r) {
        const int ma = mbase + r + 8 * half;
        qy[r] = ((float)(ma >> 6) + 0.5f) * (2.0f / 64.0f) - 1.0f;
        qx[r] = ((float)(ma & 63) + 0.5f) * (2.0f / 64.0f) - 1.0f;
        mmax[r] = -3.0e38f;
        lsum[r] = 0.0f;
    }
    v8f acc0 = {}, acc1 = {}, zero = {};

#if HAVE_TDM
    if (wave == 0) {   // prologue: fill buffer 0 for nb=0
        tdm_tile32x32((unsigned int)(K_OFF * 2), kT + (size_t)0 * CDIM + h * NHC, CDIM);
        tdm_tile32x32((unsigned int)(V_OFF * 2), vC + (size_t)(h * NHC) * HW + 0, HW);
    }
#endif

    for (int nb = 0; nb < HW; nb += 32) {
        const int buf = (nb >> 5) & 1;
#if HAVE_TDM
        if (wave == 0) __builtin_amdgcn_s_wait_tensorcnt((short)0);
        __syncthreads();   // buffer `buf` ready; previous reads of buf^1 done
        if (wave == 0 && nb + 32 < HW) {
            const int nn = nb + 32, b2 = buf ^ 1;
            tdm_tile32x32((unsigned int)((K_OFF + b2 * 1024) * 2),
                          kT + (size_t)nn * CDIM + h * NHC, CDIM);
            tdm_tile32x32((unsigned int)((V_OFF + b2 * 1024) * 2),
                          vC + (size_t)(h * NHC) * HW + nn, HW);
        }
#else
        __syncthreads();
        for (int idx = threadIdx.x; idx < 1024; idx += 128) {
            const int rr = idx >> 5, cc = idx & 31;
            smem[K_OFF + buf * 1024 + idx] = kT[(size_t)(nb + rr) * CDIM + h * NHC + cc];
            smem[V_OFF + buf * 1024 + idx] = vC[(size_t)(h * NHC + rr) * HW + nb + cc];
        }
        __syncthreads();
#endif
        const unsigned short* klds = smem + K_OFF + buf * 1024;  // [n=32][kc=32]
        const unsigned short* vlds = smem + V_OFF + buf * 1024;  // [c=32][n=32]

        float st[2][8];
#pragma unroll
        for (int t = 0; t < 2; ++t) {
            BF16Frag bk;
            bk.w = *(const u16x16*)(klds + (t * 16 + m) * 32 + half * 16);
            v8f s = __builtin_amdgcn_wmma_f32_16x16x32_bf16(
                false, aq.v, false, bk.v, (short)0, zero, false, false);
            const int na = nb + t * 16 + m;
            const float py = pos[((size_t)g * HW + na) * 2 + 0];
            const float px = pos[((size_t)g * HW + na) * 2 + 1];
#pragma unroll
            for (int r = 0; r < 8; ++r) {
                const float dy = (qy[r] - py) * 0.5f;
                const float dx = (qx[r] - px) * 0.5f;
                const float gxf = (dx + 1.0f) * 0.5f * 126.0f;
                const float gyf = (dy + 1.0f) * 0.5f * 126.0f;
                const float x0 = floorf(gxf), y0 = floorf(gyf);
                const float fx = gxf - x0, fy = gyf - y0;
                const int ix0 = (int)x0, iy0 = (int)y0;
                float bias = 0.0f;
#pragma unroll
                for (int dyi = 0; dyi < 2; ++dyi) {
                    const int iy = iy0 + dyi;
                    if (iy < 0 || iy > 126) continue;
                    const float wy = dyi ? fy : (1.0f - fy);
#pragma unroll
                    for (int dxi = 0; dxi < 2; ++dxi) {
                        const int ix = ix0 + dxi;
                        if (ix < 0 || ix > 126) continue;
                        const float wx = dxi ? fx : (1.0f - fx);
                        bias += wy * wx * bf16u_to_f32(smem[RPE_OFF + iy * 128 + ix]);
                    }
                }
                st[t][r] = s[r] * ATTN_SCALE + bias;
            }
        }
        // online softmax: each row's 16 cols live across one 16-lane half
        float pr[2][8];
#pragma unroll
        for (int r = 0; r < 8; ++r) {
            float tm = fmaxf(st[0][r], st[1][r]);
#pragma unroll
            for (int off = 1; off <= 8; off <<= 1)
                tm = fmaxf(tm, __shfl_xor(tm, off, 32));
            const float nm = fmaxf(mmax[r], tm);
            const float sc = __expf(mmax[r] - nm);
            const float p0 = __expf(st[0][r] - nm);
            const float p1 = __expf(st[1][r] - nm);
            float rs = p0 + p1;
#pragma unroll
            for (int off = 1; off <= 8; off <<= 1)
                rs += __shfl_xor(rs, off, 32);
            lsum[r] = lsum[r] * sc + rs;
            mmax[r] = nm;
            acc0[r] *= sc;
            acc1[r] *= sc;
            pr[0][r] = p0;
            pr[1][r] = p1;
        }
        // P: D-layout -> A-layout via per-wave LDS scratch (same-wave DS is in-order)
        unsigned short* pws = smem + P_OFF + wave * 512;
#pragma unroll
        for (int r = 0; r < 8; ++r) {
            const int mrow = r + 8 * half;
            pws[mrow * 32 + m]      = f32_to_bf16u(pr[0][r]);
            pws[mrow * 32 + 16 + m] = f32_to_bf16u(pr[1][r]);
        }
        asm volatile("s_wait_dscnt 0" ::: "memory");
        BF16Frag ap;
        ap.w = *(const u16x16*)(pws + m * 32 + half * 16);

        BF16Frag bv0, bv1;
        bv0.w = *(const u16x16*)(vlds + m * 32 + half * 16);          // c 0..15
        bv1.w = *(const u16x16*)(vlds + (16 + m) * 32 + half * 16);   // c 16..31
        acc0 = __builtin_amdgcn_wmma_f32_16x16x32_bf16(
            false, ap.v, false, bv0.v, (short)0, acc0, false, false);
        acc1 = __builtin_amdgcn_wmma_f32_16x16x32_bf16(
            false, ap.v, false, bv1.v, (short)0, acc1, false, false);
    }

#pragma unroll
    for (int r = 0; r < 8; ++r) {
        const int ma = mbase + r + 8 * half;
        const float inv = 1.0f / lsum[r];
        attnT[(size_t)ma * CDIM + h * NHC + m]      = f32_to_bf16u(acc0[r] * inv);
        attnT[(size_t)ma * CDIM + h * NHC + 16 + m] = f32_to_bf16u(acc1[r] * inv);
    }
}

// ---------------------------------------------------------------------------
// Host launcher
// ---------------------------------------------------------------------------
extern "C" void kernel_launch(void* const* d_in, const int* in_sizes, int n_in,
                              void* d_out, int out_size, void* d_ws, size_t ws_size,
                              hipStream_t stream) {
    const float* x     = (const float*)d_in[0];
    const float* wq    = (const float*)d_in[1];
    const float* bq    = (const float*)d_in[2];
    const float* wk    = (const float*)d_in[3];
    const float* bk    = (const float*)d_in[4];
    const float* wv    = (const float*)d_in[5];
    const float* bv    = (const float*)d_in[6];
    const float* w_dw  = (const float*)d_in[7];
    const float* b_dw  = (const float*)d_in[8];
    const float* ln_g  = (const float*)d_in[9];
    const float* ln_b  = (const float*)d_in[10];
    const float* w_pw  = (const float*)d_in[11];
    const float* rpe   = (const float*)d_in[12];
    const float* w_out = (const float*)d_in[13];
    const float* b_out = (const float*)d_in[14];
    (void)in_sizes; (void)n_in; (void)out_size; (void)ws_size;

    char* wsp = (char*)d_ws;
    size_t off = 0;
    auto alloc = [&](size_t bytes) -> void* {
        void* p = wsp + off;
        off = (off + bytes + 255) & ~(size_t)255;
        return p;
    };
    const size_t NCS = (size_t)CDIM * HW;
    unsigned short* xT    = (unsigned short*)alloc(NCS * 2);   // x token-major bf16
    unsigned short* wqb   = (unsigned short*)alloc(65536 * 2);
    unsigned short* wkb   = (unsigned short*)alloc(65536 * 2);
    unsigned short* wvb   = (unsigned short*)alloc(65536 * 2);
    unsigned short* wob   = (unsigned short*)alloc(65536 * 2);
    float*          qf    = (float*)alloc(NCS * 4);            // q channel-major f32
    unsigned short* qTb   = (unsigned short*)alloc(NCS * 2);   // q token-major bf16
    float*          odw   = (float*)alloc(NCS * 4);
    float*          posb  = (float*)alloc((size_t)NGRP * HW * 2 * 4);
    unsigned short* xsT   = (unsigned short*)alloc(NCS * 2);   // sampled token-major
    unsigned short* kTb   = (unsigned short*)alloc(NCS * 2);   // k token-major
    unsigned short* vCb   = (unsigned short*)alloc(NCS * 2);   // v channel-major
    unsigned short* attnT = (unsigned short*)alloc(NCS * 2);   // attn out token-major

    f2bf_tr_kernel<<<4096, 256, 0, stream>>>(x, xT);
    f2bf_kernel<<<256, 256, 0, stream>>>(wq, wqb, 65536);
    f2bf_kernel<<<256, 256, 0, stream>>>(wk, wkb, 65536);
    f2bf_kernel<<<256, 256, 0, stream>>>(wv, wvb, 65536);
    f2bf_kernel<<<256, 256, 0, stream>>>(w_out, wob, 65536);

    // q projection: f32 channel-major (offset branch) + bf16 token-major (attn)
    gemm_bias_kernel<<<1024, 128, 0, stream>>>(wqb, xT, bq, qf, qTb, nullptr, HW);

    // offset branch
    dwconv_kernel<<<4096, 256, 0, stream>>>(qf, w_dw, b_dw, odw);
    lnpw_kernel<<<64, 256, 0, stream>>>(odw, ln_g, ln_b, w_pw, posb);

    // deformable sampling + k/v projections
    sample_kernel<<<4096, 256, 0, stream>>>(x, posb, xsT);
    gemm_bias_kernel<<<1024, 128, 0, stream>>>(wkb, xsT, bk, nullptr, kTb, nullptr, HW);
    gemm_bias_kernel<<<1024, 128, 0, stream>>>(wvb, xsT, bv, nullptr, nullptr, vCb, HW);

    // fused flash attention (TDM-staged K/V tiles, LDS RPE)
    attn_kernel<<<dim3(64, 8), 128, 0, stream>>>(qTb, kTb, vCb, posb, rpe, attnT);

    // output projection -> d_out f32 [1,256,64,64]
    gemm_bias_kernel<<<1024, 128, 0, stream>>>(wob, attnT, b_out,
                                               (float*)d_out, nullptr, nullptr, HW);
}
